// ReacHGNN_8744553414948
// MI455X (gfx1250) — matrix operand order repaired and checked by reading, not compiled, not run on hardware
//
#include <hip/hip_runtime.h>
#include <hip/hip_bf16.h>
#include <math.h>

typedef __attribute__((ext_vector_type(16))) _Float16 v16h;
typedef __attribute__((ext_vector_type(8)))  float    v8f;
typedef unsigned int u32;
typedef __attribute__((ext_vector_type(4)))  u32      u32x4;

#define NN   2048
#define EE   2048
#define EICNT 8192
#define BGR  64
#define XINF 64
#define HF   128
#define OHF  512
#define INV_BN 0.9999950000374997f

__device__ __forceinline__ float sigf(float x){ return 1.f/(1.f+expf(-x)); }
__device__ __forceinline__ u32 encf(float f){ int i=__float_as_int(f); return (i>=0)?((u32)i|0x80000000u):~((u32)i); }
__device__ __forceinline__ float decf(u32 u){ int i=(u&0x80000000u)?(int)(u&0x7fffffffu):~(int)u; return __int_as_float(i); }

union h2u { u32 u; _Float16 h[2]; };

// -------- generic pack: weight (f32) -> WMMA-B fragment-ordered f16 tiles ----------
// out layout: [(kt*Ntiles + nt)*32 + lane][16] ; per-lane K map = ((v&4)?16:0) + (laneHi?8:0) + 2(v&3) (+j)
// trans=0: src is row-major [Ksrc, N], srcLd = N ; trans=1: src is [N, Ksrc], srcLd = Ksrc
__global__ void k_pack_b(const float* __restrict__ src, int Ksrc, int Ktiles, int Ntiles,
                         int srcLd, int trans, _Float16* __restrict__ out){
  int idx = blockIdx.x*blockDim.x + threadIdx.x;
  if (idx >= Ktiles*Ntiles*32) return;
  int lane = idx & 31; int t = idx >> 5;
  int nt = t % Ntiles; int kt = t / Ntiles;
  int hf = lane >> 4, l15 = lane & 15;
  int n = nt*16 + l15;
  _Float16* o = out + (size_t)idx*16;
#pragma unroll
  for (int v = 0; v < 8; ++v){
    int kk = kt*32 + ((v&4)?16:0) + (hf?8:0) + 2*(v&3);
    float f0 = 0.f, f1 = 0.f;
    if (kk < Ksrc)     f0 = trans ? src[(size_t)n*srcLd + kk]     : src[(size_t)kk*srcLd + n];
    if (kk + 1 < Ksrc) f1 = trans ? src[(size_t)n*srcLd + kk + 1] : src[(size_t)(kk+1)*srcLd + n];
    o[2*v]   = (_Float16)f0;
    o[2*v+1] = (_Float16)f1;
  }
}

// ---------------- templated WMMA GEMM: C[M,N] = act(A_f16[M,K] @ packB + bias) -----
template<int K>
__global__ void __launch_bounds__(32)
k_gemm_t(const _Float16* __restrict__ A, int lda,
         const v16h* __restrict__ Bp,
         const float* __restrict__ bias,
         float* __restrict__ C, int ldc, int act)
{
  const int lane = threadIdx.x;
  const int hf = lane >> 4, l15 = lane & 15;
  const int row = blockIdx.y*16 + l15;
  const int col = blockIdx.x*16 + l15;
  const u32* Au = (const u32*)(A + (size_t)row*lda);
  const v16h* bptr = Bp + (size_t)blockIdx.x*32 + lane;
  const int bstride = gridDim.x*32;
  v8f acc = {};
  for (int kt = 0; kt < K/32; ++kt){
    v16h a;
#pragma unroll
    for (int v = 0; v < 8; ++v){
      int kaA = kt*32 + ((v&4)?16:0) + (hf?8:0) + (v&3)*2;  // even
      h2u cv; cv.u = Au[kaA >> 1];
      a[2*v] = cv.h[0]; a[2*v+1] = cv.h[1];
    }
    v16h b = bptr[(size_t)kt*bstride];
    acc = __builtin_amdgcn_wmma_f32_16x16x32_f16(false, a, false, b, (short)0, acc, false, false);
  }
#pragma unroll
  for (int r = 0; r < 8; ++r){
    int rr = blockIdx.y*16 + r + (hf?8:0);
    float v = acc[r];
    if (bias) v += bias[col];
    if (act == 1) v = v > 0.f ? v : 0.f;
    C[(size_t)rr*ldc + col] = v;
  }
}

// ------------- fused NNConv message: msg[e,:] = (h1[e] x x[src e]) @ w2 -------------
// block = 256 threads (8 waves), 64 edges, N = 128, K = 32768 (k-major, i-minor).
// Wave w&3 -> edge tile (16 edges); wave w>>2 -> half of the 8 N-tiles. Each wave
// carries only 4 accumulators (32 VGPRs) to avoid acc live-range splitting.
// B tile (8 KB) staged to LDS with register double-buffering: next tile's 2x b128
// global loads issued before the WMMAs, stored to the alternate LDS buffer after.
__global__ void __launch_bounds__(256)
k_fused_msg(const _Float16* __restrict__ x16, const _Float16* __restrict__ h1,
            const _Float16* __restrict__ pack, const int* __restrict__ ei,
            float* __restrict__ msg)
{
  __shared__ _Float16 xsh[64][128];
  __shared__ _Float16 h1sh[64][256];
  __shared__ v16h bsh[2][8][32];
  const int tid = threadIdx.x, lane = tid & 31, wave = tid >> 5;
  const int et = wave & 3;                        // edge tile 0..3
  const int nh = wave >> 2;                       // N half: tiles nh*4 .. nh*4+3
  const int eBase = blockIdx.x * 64;
  for (int idx = tid; idx < 64*128; idx += 256){
    int el = idx >> 7, c = idx & 127;
    int s = ei[eBase + el];                       // ei row0 = src
    xsh[el][c] = x16[(size_t)s*128 + c];
  }
  for (int idx = tid; idx < 64*256; idx += 256){
    int el = idx >> 8, c = idx & 255;
    h1sh[el][c] = h1[(size_t)(eBase + el)*256 + c];
  }
  const int hf = lane >> 4, l15 = lane & 15;
  const int eloc = et*16 + l15;
  v8f zero = {};
  v8f acc[4];
#pragma unroll
  for (int i = 0; i < 4; ++i) acc[i] = zero;
  const u32* xrow = (const u32*)&xsh[eloc][0];
  const u32x4* gsrc = (const u32x4*)pack;         // 512 x u32x4 per K-tile
  u32x4* lbase = (u32x4*)&bsh[0][0][0];           // 512 x u32x4 per LDS buffer
  u32x4 stg[2];
  // prologue: stage tile 0 into buffer 0
#pragma unroll
  for (int q = 0; q < 2; ++q) stg[q] = gsrc[(size_t)tid*2 + q];
#pragma unroll
  for (int q = 0; q < 2; ++q) lbase[tid*2 + q] = stg[q];
  __syncthreads();
  for (int kt = 0; kt < 1024; ++kt){
    const int cur = kt & 1;
    const bool more = (kt + 1) < 1024;
    if (more){
      const u32x4* gp = gsrc + (size_t)(kt + 1)*512 + tid*2;
#pragma unroll
      for (int q = 0; q < 2; ++q) stg[q] = gp[q];
      __builtin_prefetch(gsrc + (size_t)(kt + 2)*512, 0, 1);  // global_prefetch_b8
    }
    const int kc = kt >> 2;                       // which h1 column this 32-window uses
    const int i0 = (kt & 3) * 32;                 // x column base
    _Float16 hs = h1sh[eloc][kc];
    v16h a;
#pragma unroll
    for (int v = 0; v < 8; ++v){
      int kb = ((v&4)?16:0) + (hf?8:0) + (v&3)*2; // even
      h2u cv; cv.u = xrow[(i0 + kb) >> 1];
      a[2*v]   = hs * cv.h[0];
      a[2*v+1] = hs * cv.h[1];
    }
#pragma unroll
    for (int nt = 0; nt < 4; ++nt){
      v16h b = bsh[cur][nh*4 + nt][lane];
      acc[nt] = __builtin_amdgcn_wmma_f32_16x16x32_f16(false, a, false, b, (short)0, acc[nt], false, false);
    }
    if (more){
      u32x4* lb = lbase + (size_t)(cur ^ 1)*512 + tid*2;
#pragma unroll
      for (int q = 0; q < 2; ++q) lb[q] = stg[q];
    }
    __syncthreads();
  }
#pragma unroll
  for (int nt = 0; nt < 4; ++nt)
#pragma unroll
    for (int r = 0; r < 8; ++r){
      int e = eBase + et*16 + r + (hf?8:0);
      msg[(size_t)e*128 + (nh*4 + nt)*16 + l15] = acc[nt][r];
    }
}

// ------------------------------ elementwise helpers --------------------------------
__global__ void k_cvt(const float* __restrict__ s, _Float16* __restrict__ d, int n){
  int i = blockIdx.x*blockDim.x + threadIdx.x; if (i < n) d[i] = (_Float16)s[i];
}
__global__ void k_cvt_pad(const float* __restrict__ s, _Float16* __restrict__ d,
                          int rows, int cs, int cd){
  int i = blockIdx.x*blockDim.x + threadIdx.x; if (i >= rows*cd) return;
  int r = i / cd, c = i % cd;
  d[i] = (c < cs) ? (_Float16)s[(size_t)r*cs + c] : (_Float16)0.f;
}
__global__ void k_copy(const float* __restrict__ s, float* __restrict__ d, int n){
  int i = blockIdx.x*blockDim.x + threadIdx.x; if (i < n) d[i] = s[i];
}
__global__ void k_fill_u32(u32* p, u32 v, int n){
  int i = blockIdx.x*blockDim.x + threadIdx.x; if (i < n) p[i] = v;
}
__global__ void k_count(const int* __restrict__ dst, float* __restrict__ cnt, int e){
  int i = blockIdx.x*blockDim.x + threadIdx.x; if (i < e) atomicAdd(&cnt[dst[i]], 1.f);
}
__global__ void k_scatter_msg(const float* __restrict__ msg, const float* __restrict__ nb,
                              const int* __restrict__ ei, float* __restrict__ sagg, int e){
  int idx = blockIdx.x*blockDim.x + threadIdx.x; if (idx >= e*128) return;
  int ed = idx >> 7, c = idx & 127;
  int s = ei[ed], d = ei[e + ed];
  atomicAdd(&sagg[(size_t)d*128 + c], msg[idx] + nb[(size_t)s*128 + c]);
}
__global__ void k_nnconv_combine(const float* __restrict__ root, const float* __restrict__ sagg,
                                 const float* __restrict__ cnt, const float* __restrict__ bias,
                                 float* __restrict__ out){
  int idx = blockIdx.x*blockDim.x + threadIdx.x; if (idx >= NN*128) return;
  int n = idx >> 7, c = idx & 127;
  float cn = cnt[n]; cn = cn > 1.f ? cn : 1.f;
  float v = root[idx] + sagg[idx]/cn + bias[c];
  out[idx] = v > 0.f ? v : 0.f;                  // ReLU of NNConv output
}
__global__ void k_gru_gates(const float* __restrict__ gi, const float* __restrict__ gh,
                            const float* __restrict__ hp, float* __restrict__ o1,
                            float* __restrict__ o2){
  int idx = blockIdx.x*blockDim.x + threadIdx.x; if (idx >= NN*128) return;
  int n = idx >> 7, c = idx & 127;
  const float* gin = gi + (size_t)n*384; const float* ghn = gh + (size_t)n*384;
  float r = sigf(gin[c] + ghn[c]);
  float z = sigf(gin[128+c] + ghn[128+c]);
  float nn = tanhf(gin[256+c] + r*ghn[256+c]);
  float h = (1.f - z)*nn + z*hp[idx];
  o1[idx] = h; if (o2) o2[idx] = h;
}
__global__ void k_gat_logits(const float* __restrict__ xl, const float* __restrict__ xr,
                             const float* __restrict__ att, const int* __restrict__ ei,
                             float* __restrict__ logit, u32* __restrict__ nmax){
  int idx = blockIdx.x*blockDim.x + threadIdx.x; if (idx >= EICNT*2) return;
  int e = idx >> 1, h = idx & 1;
  int s = ei[e], d = ei[EICNT + e];
  float acc = 0.f;
  for (int c = 0; c < 64; ++c){
    float v = xl[(size_t)s*128 + h*64 + c] + xr[(size_t)d*128 + h*64 + c];
    v = v > 0.f ? v : 0.2f*v;                    // leaky_relu 0.2
    acc += v * att[h*64 + c];
  }
  logit[idx] = acc;
  atomicMax(&nmax[d*2 + h], encf(acc));
}
__global__ void k_gat_exp(const float* __restrict__ logit, const u32* __restrict__ nmax,
                          const int* __restrict__ ei, float* __restrict__ ev,
                          float* __restrict__ den){
  int idx = blockIdx.x*blockDim.x + threadIdx.x; if (idx >= EICNT*2) return;
  int e = idx >> 1, h = idx & 1;
  int d = ei[EICNT + e];
  float m = decf(nmax[d*2 + h]); if (!isfinite(m)) m = 0.f;
  float x = expf(logit[idx] - m);
  ev[idx] = x;
  atomicAdd(&den[d*2 + h], x);
}
__global__ void k_gat_scatter(const float* __restrict__ xl, const float* __restrict__ ev,
                              const float* __restrict__ den, const int* __restrict__ ei,
                              float* __restrict__ gout){
  int idx = blockIdx.x*blockDim.x + threadIdx.x; if (idx >= EICNT*128) return;
  int e = idx >> 7, c = idx & 127, h = c >> 6;
  int s = ei[e], d = ei[EICNT + e];
  float a = ev[e*2 + h] / (den[d*2 + h] + 1e-16f);
  atomicAdd(&gout[(size_t)d*128 + c], xl[(size_t)s*128 + c] * a);
}
__global__ void k_gat_combine(const float* __restrict__ xs, const float* __restrict__ g0,
                              const float* __restrict__ g1, const float* __restrict__ b0,
                              const float* __restrict__ b1, float* __restrict__ out){
  int idx = blockIdx.x*blockDim.x + threadIdx.x; if (idx >= NN*128) return;
  int c = idx & 127;
  out[idx] = xs[idx] + g0[idx] + b0[c] + g1[idx] + b1[c];
}
__global__ void __launch_bounds__(128)
k_graphnorm(const float* __restrict__ x, const float* __restrict__ gw,
            const float* __restrict__ gb, const float* __restrict__ gms,
            float* __restrict__ y){
  int b = blockIdx.x, c = threadIdx.x;
  const float* xb = x + (size_t)b*32*128;
  float m = 0.f;
  for (int r = 0; r < 32; ++r) m += xb[r*128 + c];
  m *= (1.f/32.f);
  float ms = gms[c];
  float var = 0.f;
  for (int r = 0; r < 32; ++r){ float o = xb[r*128 + c] - m*ms; var += o*o; }
  var *= (1.f/32.f);
  float inv = rsqrtf(var + 1e-5f);
  float w = gw[c], bb = gb[c];
  float* yb = y + (size_t)b*32*128;
  for (int r = 0; r < 32; ++r){ float o = xb[r*128 + c] - m*ms; yb[r*128 + c] = w*o*inv + bb; }
}
__global__ void k_outs1(const float* __restrict__ xr1, const float* __restrict__ xr2,
                        const float* __restrict__ xp, const int* __restrict__ reac,
                        const int* __restrict__ prod, float* __restrict__ o){
  int idx = blockIdx.x*blockDim.x + threadIdx.x; if (idx >= BGR*512) return;
  int b = idx >> 9, c = idx & 511;
  int seg = c >> 7, cc = c & 127;
  const float* src; int row;
  if      (seg == 0){ row = reac[b*2+0]; src = xr1; }
  else if (seg == 1){ row = prod[b*2+0]; src = xp;  }
  else if (seg == 2){ row = reac[b*2+1]; src = xr2; }
  else              { row = prod[b*2+1]; src = xp;  }
  o[idx] = src[(size_t)row*128 + cc];
}
__global__ void __launch_bounds__(128)
k_segmax(const float* __restrict__ x, float* __restrict__ out, int ld, int coff){
  int b = blockIdx.x, c = threadIdx.x;
  float m = -INFINITY;
  for (int r = 0; r < 32; ++r) m = fmaxf(m, x[(size_t)(b*32+r)*128 + c]);
  out[(size_t)b*ld + coff + c] = m;
}
__global__ void __launch_bounds__(128)
k_segmean(const float* __restrict__ x, float* __restrict__ out, int ld, int coff){
  int b = blockIdx.x, c = threadIdx.x;
  float s = 0.f;
  for (int r = 0; r < 32; ++r) s += x[(size_t)(b*32+r)*128 + c];
  out[(size_t)b*ld + coff + c] = s * (1.f/32.f);
}
__global__ void __launch_bounds__(128)
k_set2set(const float* __restrict__ x, const float* __restrict__ wih,
          const float* __restrict__ whh, const float* __restrict__ bih,
          const float* __restrict__ bhh, float* __restrict__ out, int ld, int coff){
  __shared__ float xsh[32*128];
  __shared__ float hs[128], cs[128], qstar[256], g[512], att[32], rr[128];
  int b = blockIdx.x, tid = threadIdx.x;
  for (int i = tid; i < 32*128; i += 128) xsh[i] = x[(size_t)(b*32)*128 + i];
  hs[tid] = 0.f; cs[tid] = 0.f; qstar[tid] = 0.f; qstar[128+tid] = 0.f;
  __syncthreads();
  for (int it = 0; it < 3; ++it){
    for (int j = tid; j < 512; j += 128){
      float acc = bih[j] + bhh[j];
      for (int k = 0; k < 256; ++k) acc += qstar[k]*wih[(size_t)j*256 + k];
      for (int k = 0; k < 128; ++k) acc += hs[k]*whh[(size_t)j*128 + k];
      g[j] = acc;
    }
    __syncthreads();
    {
      float c = sigf(g[128+tid])*cs[tid] + sigf(g[tid])*tanhf(g[256+tid]);
      cs[tid] = c;
      hs[tid] = sigf(g[384+tid])*tanhf(c);
    }
    __syncthreads();
    if (tid < 32){
      float acc = 0.f;
      for (int c = 0; c < 128; ++c) acc += xsh[tid*128 + c]*hs[c];
      att[tid] = acc;
    }
    __syncthreads();
    if (tid == 0){
      float m = -INFINITY;
      for (int n = 0; n < 32; ++n) m = fmaxf(m, att[n]);
      if (!isfinite(m)) m = 0.f;
      float s = 0.f;
      for (int n = 0; n < 32; ++n){ att[n] = expf(att[n]-m); s += att[n]; }
      float inv = 1.f/(s + 1e-16f);
      for (int n = 0; n < 32; ++n) att[n] *= inv;
    }
    __syncthreads();
    { float acc = 0.f; for (int n = 0; n < 32; ++n) acc += att[n]*xsh[n*128 + tid]; rr[tid] = acc; }
    __syncthreads();
    qstar[tid] = hs[tid]; qstar[128+tid] = rr[tid];
    __syncthreads();
  }
  out[(size_t)b*ld + coff + tid]       = qstar[tid];
  out[(size_t)b*ld + coff + 128 + tid] = qstar[128+tid];
}
__global__ void k_bnlin(float* __restrict__ y, int ld, const float* __restrict__ g,
                        const float* __restrict__ be){
  int idx = blockIdx.x*blockDim.x + threadIdx.x; if (idx >= BGR*512) return;
  int r = idx >> 9, c = idx & 511;
  float* p = y + (size_t)r*ld + c;
  float v = (*p) * INV_BN * g[c] + be[c];
  *p = v * sigf(v);                              // swish
}
__global__ void k_prelu(float* __restrict__ y, const float* __restrict__ a, int n){
  int i = blockIdx.x*blockDim.x + threadIdx.x; if (i >= n) return;
  float v = y[i]; float av = a[0];
  y[i] = v > 0.f ? v : av*v;
}
__global__ void __launch_bounds__(128)
k_final(const float* __restrict__ y2, const float* __restrict__ w3,
        const float* __restrict__ b3, float* __restrict__ out){
  __shared__ float s0[128], s1[128];
  int b = blockIdx.x, tid = threadIdx.x;
  float a0 = 0.f, a1 = 0.f;
  for (int k = tid; k < 512; k += 128){
    float v = y2[(size_t)b*512 + k];
    a0 += v*w3[k*2+0]; a1 += v*w3[k*2+1];
  }
  s0[tid] = a0; s1[tid] = a1; __syncthreads();
  for (int off = 64; off > 0; off >>= 1){
    if (tid < off){ s0[tid] += s0[tid+off]; s1[tid] += s1[tid+off]; }
    __syncthreads();
  }
  if (tid == 0){ out[b] = s0[0] + b3[0]; out[64 + b] = s1[0] + b3[1]; }
}

// ===================================== host ========================================
extern "C" void kernel_launch(void* const* d_in, const int* in_sizes, int n_in,
                              void* d_out, int out_size, void* d_ws, size_t ws_size,
                              hipStream_t stream)
{
  (void)in_sizes; (void)n_in; (void)out_size; (void)ws_size;
  const float* x0[3] = {(const float*)d_in[0], (const float*)d_in[1], (const float*)d_in[2]};
  const float* ea[3] = {(const float*)d_in[3], (const float*)d_in[4], (const float*)d_in[5]};
  // params flattened in _make_params insertion order starting at index 6
  auto F = [&](int i)->const float*{ return (const float*)d_in[6 + i]; };
  const float *lin0_w=F(0), *lin0_b=F(1), *en_w1=F(2), *en_b1=F(3), *en_w2=F(4), *en_b2=F(5);
  const float *conv_root=F(6), *conv_bias=F(7);
  const float *gru_wih=F(8), *gru_whh=F(9), *gru_bih=F(10), *gru_bhh=F(11);
  const float *gn_w=F(12), *gn_b=F(13), *gn_ms=F(14);
  const float *s2s_wih=F(15), *s2s_whh=F(16), *s2s_bih=F(17), *s2s_bhh=F(18);
  const float *gat_ll=F(19), *gat_lr=F(20), *gat_att=F(21), *gat_bp=F(22);
  const float *bl_w[4]={F(23),F(27),F(31),F(35)}, *bl_b[4]={F(24),F(28),F(32),F(36)};
  const float *bl_g[4]={F(25),F(29),F(33),F(37)}, *bl_be[4]={F(26),F(30),F(34),F(38)};
  const float *out_w1=F(39), *out_b1=F(40), *out_a1=F(41);
  const float *out_w2=F(42), *out_b2=F(43), *out_a2=F(44);
  const float *out_w3=F(45), *out_b3=F(46);
  const int* eiG[3]  = {(const int*)d_in[53], (const int*)d_in[54], (const int*)d_in[55]}; // r1,r2,p
  const int* ei_r1p  = (const int*)d_in[56];
  const int* ei_pr1  = (const int*)d_in[57];
  const int* ei_r2p  = (const int*)d_in[58];
  const int* ei_pr2  = (const int*)d_in[59];
  const int* ei_r1r2 = (const int*)d_in[60];
  const int* ei_r2r1 = (const int*)d_in[61];
  const int* reac    = (const int*)d_in[65];
  const int* prodS   = (const int*)d_in[66];

  char* wp = (char*)d_ws;
  auto alloc = [&](size_t bytes)->char*{ char* p = wp; wp += (bytes + 255) & ~(size_t)255; return p; };
  _Float16* w2pack  = (_Float16*)alloc((size_t)32768*128*2);
  _Float16* enw1p   = (_Float16*)alloc((size_t)32*256*2);
  _Float16* lin0p   = (_Float16*)alloc((size_t)XINF*HF*2);
  _Float16* b2mp    = (_Float16*)alloc((size_t)HF*HF*2);
  _Float16* crootp  = (_Float16*)alloc((size_t)HF*HF*2);
  _Float16* wihp    = (_Float16*)alloc((size_t)HF*384*2);
  _Float16* whhp    = (_Float16*)alloc((size_t)HF*384*2);
  _Float16* gllp    = (_Float16*)alloc((size_t)6*HF*HF*2);
  _Float16* glrp    = (_Float16*)alloc((size_t)6*HF*HF*2);
  _Float16* blwp[4];
  int blK[4] = {512, 384, 384, 768};
  for (int i = 0; i < 4; ++i) blwp[i] = (_Float16*)alloc((size_t)blK[i]*512*2);
  _Float16* ow1p = (_Float16*)alloc((size_t)2048*512*2);
  _Float16* ow2p = (_Float16*)alloc((size_t)512*512*2);
  float *xsP[3], *hsP[3], *xnP[3], *xgP[3], *cntP[3];
  _Float16 *x16P[3], *h1P[3];
  for (int g = 0; g < 3; ++g){
    xsP[g] = (float*)alloc((size_t)NN*HF*4);  hsP[g] = (float*)alloc((size_t)NN*HF*4);
    xnP[g] = (float*)alloc((size_t)NN*HF*4);  xgP[g] = (float*)alloc((size_t)NN*HF*4);
    x16P[g] = (_Float16*)alloc((size_t)NN*HF*2);
    h1P[g]  = (_Float16*)alloc((size_t)EE*256*2);
    cntP[g] = (float*)alloc(NN*4);
  }
  _Float16* ea16   = (_Float16*)alloc((size_t)EE*32*2);    // padded K=16 -> 32
  _Float16* xin16  = (_Float16*)alloc((size_t)NN*XINF*2);
  float* h1f    = (float*)alloc((size_t)EE*256*4);
  float* nodeB  = (float*)alloc((size_t)NN*HF*4);
  float* rootO  = (float*)alloc((size_t)NN*HF*4);
  float* sagg   = (float*)alloc((size_t)NN*HF*4);
  float* msg    = (float*)alloc((size_t)EE*HF*4);
  float* mbuf   = (float*)alloc((size_t)NN*HF*4);
  _Float16* m16 = (_Float16*)alloc((size_t)NN*HF*2);
  _Float16* hg16= (_Float16*)alloc((size_t)NN*HF*2);
  float* gi     = (float*)alloc((size_t)NN*384*4);
  float* gh     = (float*)alloc((size_t)NN*384*4);
  float* xl     = (float*)alloc((size_t)NN*HF*4);
  float* xr     = (float*)alloc((size_t)NN*HF*4);
  float* glog   = (float*)alloc((size_t)EICNT*2*4);
  float* gexp   = (float*)alloc((size_t)EICNT*2*4);
  u32*   nmax   = (u32*)alloc((size_t)NN*2*4);
  float* gden   = (float*)alloc((size_t)NN*2*4);
  float* gout0  = (float*)alloc((size_t)NN*HF*4);
  float* gout1  = (float*)alloc((size_t)NN*HF*4);
  float* outs1  = (float*)alloc((size_t)BGR*512*4);
  float* outs2  = (float*)alloc((size_t)BGR*384*4);
  float* outs3  = (float*)alloc((size_t)BGR*384*4);
  float* outs4  = (float*)alloc((size_t)BGR*768*4);
  float* ycat   = (float*)alloc((size_t)BGR*2048*4);
  _Float16* o16 = (_Float16*)alloc((size_t)BGR*2048*2);
  float* y1     = (float*)alloc((size_t)BGR*512*4);
  float* y2     = (float*)alloc((size_t)BGR*512*4);

  auto cvt = [&](const float* s, _Float16* d, int n){
    k_cvt<<<(n+255)/256, 256, 0, stream>>>(s, d, n);
  };
  auto fill0 = [&](void* p, int n){          // n u32 words
    k_fill_u32<<<(n+255)/256, 256, 0, stream>>>((u32*)p, 0u, n);
  };
  auto pack = [&](const float* src, int Ksrc, int Kpad, int Nn, int srcLd, int trans,
                  _Float16* out){
    int tot = (Kpad/32)*(Nn/16)*32;
    k_pack_b<<<(tot+255)/256, 256, 0, stream>>>(src, Ksrc, Kpad/32, Nn/16, srcLd, trans, out);
  };
  auto gemm = [&](int K, const _Float16* A, int lda, const _Float16* Bp, const float* bias,
                  float* C, int ldc, int M, int Nn, int act){
    dim3 gr(Nn/16, M/16);
    const v16h* B = (const v16h*)Bp;
    switch (K){
      case 32:   k_gemm_t<32>  <<<gr,32,0,stream>>>(A,lda,B,bias,C,ldc,act); break;
      case 64:   k_gemm_t<64>  <<<gr,32,0,stream>>>(A,lda,B,bias,C,ldc,act); break;
      case 128:  k_gemm_t<128> <<<gr,32,0,stream>>>(A,lda,B,bias,C,ldc,act); break;
      case 384:  k_gemm_t<384> <<<gr,32,0,stream>>>(A,lda,B,bias,C,ldc,act); break;
      case 512:  k_gemm_t<512> <<<gr,32,0,stream>>>(A,lda,B,bias,C,ldc,act); break;
      case 768:  k_gemm_t<768> <<<gr,32,0,stream>>>(A,lda,B,bias,C,ldc,act); break;
      case 2048: k_gemm_t<2048><<<gr,32,0,stream>>>(A,lda,B,bias,C,ldc,act); break;
      default: break;
    }
  };

  // ---- one-time (per call) weight prep: pack all B operands to fragment order ----
  pack(en_w2, 32768, 32768, 128, 128, 0, w2pack);   // huge edge-MLP weight (L2-resident)
  pack(en_w1, 16, 32, 256, 256, 0, enw1p);          // zero-padded K 16->32
  pack(lin0_w, XINF, XINF, HF, HF, 0, lin0p);
  pack(en_b2, HF, HF, HF, HF, 0, b2mp);             // en_b2 viewed as [128,128]
  pack(conv_root, HF, HF, HF, HF, 0, crootp);
  pack(gru_wih, HF, HF, 384, HF, 1, wihp);          // used as m @ wih.T
  pack(gru_whh, HF, HF, 384, HF, 1, whhp);
  for (int t = 0; t < 6; ++t){
    pack(gat_ll + (size_t)t*HF*HF, HF, HF, HF, HF, 0, gllp + (size_t)t*HF*HF);
    pack(gat_lr + (size_t)t*HF*HF, HF, HF, HF, HF, 0, glrp + (size_t)t*HF*HF);
  }
  for (int i = 0; i < 4; ++i) pack(bl_w[i], blK[i], blK[i], 512, 512, 0, blwp[i]);
  pack(out_w1, 2048, 2048, 512, 512, 0, ow1p);
  pack(out_w2, 512, 512, 512, 512, 0, ow2p);

  // ---- per-graph edge MLP (depends only on ea) + in-degree counts ----
  for (int g = 0; g < 3; ++g){
    k_cvt_pad<<<(EE*32+255)/256, 256, 0, stream>>>(ea[g], ea16, EE, 16, 32);
    gemm(32, ea16, 32, enw1p, en_b1, h1f, 256, EE, 256, 1);   // relu
    cvt(h1f, h1P[g], EE*256);
    fill0(cntP[g], NN);
    k_count<<<EE/256, 256, 0, stream>>>(eiG[g] + EE, cntP[g], EE);
  }

  auto nnconv = [&](int g){
    cvt(xsP[g], x16P[g], NN*HF);
    gemm(128, x16P[g], HF, b2mp, nullptr, nodeB, HF, NN, HF, 0);   // x @ en_b2-matrix
    gemm(128, x16P[g], HF, crootp, nullptr, rootO, HF, NN, HF, 0); // x @ conv_root
    fill0(sagg, NN*HF);
    k_fused_msg<<<EE/64, 256, 0, stream>>>(x16P[g], h1P[g], w2pack, eiG[g], msg);
    k_scatter_msg<<<(EE*128)/256, 256, 0, stream>>>(msg, nodeB, eiG[g], sagg, EE);
    k_nnconv_combine<<<(NN*128)/256, 256, 0, stream>>>(rootO, sagg, cntP[g], conv_bias, mbuf);
  };
  auto gru = [&](int g){
    cvt(mbuf, m16, NN*HF);
    cvt(hsP[g], hg16, NN*HF);
    gemm(128, m16, HF, wihp, gru_bih, gi, 384, NN, 384, 0);  // m @ wih.T
    gemm(128, hg16, HF, whhp, gru_bhh, gh, 384, NN, 384, 0); // h @ whh.T
    k_gru_gates<<<(NN*128)/256, 256, 0, stream>>>(gi, gh, hsP[g], xsP[g], hsP[g]);
  };

  // ---- pre-MPNN ----
  for (int g = 0; g < 3; ++g){
    cvt(x0[g], xin16, NN*XINF);
    gemm(64, xin16, XINF, lin0p, lin0_b, xsP[g], HF, NN, HF, 1); // relu
    k_copy<<<(NN*128)/256, 256, 0, stream>>>(xsP[g], hsP[g], NN*HF);
    for (int it = 0; it < 2; ++it){ nnconv(g); gru(g); }
  }

  // ---- cross-graph blocks ----
  auto gat_pair = [&](int tg, int sA, int tA, const int* eiA,
                      int sB, int tB, const int* eiB, float* dst){
    int ts[2] = {tA, tB}; int ss[2] = {sA, sB}; const int* es[2] = {eiA, eiB};
    float* go[2] = {gout0, gout1};
    for (int j = 0; j < 2; ++j){
      int t = ts[j];
      gemm(128, x16P[ss[j]], HF, gllp + (size_t)t*HF*HF, nullptr, xl, HF, NN, HF, 0);
      gemm(128, x16P[tg],    HF, glrp + (size_t)t*HF*HF, nullptr, xr, HF, NN, HF, 0);
      k_fill_u32<<<(NN*2+255)/256, 256, 0, stream>>>(nmax, 0x007FFFFFu, NN*2); // enc(-inf)
      fill0(gden, NN*2); fill0(go[j], NN*HF);
      k_gat_logits<<<(EICNT*2)/256, 256, 0, stream>>>(xl, xr, gat_att + t*HF, es[j], glog, nmax);
      k_gat_exp<<<(EICNT*2)/256, 256, 0, stream>>>(glog, nmax, es[j], gexp, gden);
      k_gat_scatter<<<(EICNT*128)/256, 256, 0, stream>>>(xl, gexp, gden, es[j], go[j]);
    }
    k_gat_combine<<<(NN*128)/256, 256, 0, stream>>>(xsP[tg], gout0, gout1,
                                                    gat_bp + tA*HF, gat_bp + tB*HF, dst);
  };
  for (int blk = 0; blk < 2; ++blk){
    for (int g = 0; g < 3; ++g) cvt(xsP[g], x16P[g], NN*HF);
    gat_pair(2, 0, 0, ei_r1p, 1, 1, ei_r2p, xnP[2]);   // d_p
    gat_pair(0, 2, 2, ei_pr1, 1, 3, ei_r2r1, xnP[0]);  // d_r1
    gat_pair(1, 2, 4, ei_pr2, 0, 5, ei_r1r2, xnP[1]);  // d_r2
    for (int g = 0; g < 3; ++g){ float* t = xsP[g]; xsP[g] = xnP[g]; xnP[g] = t; }
    for (int g = 0; g < 3; ++g){ nnconv(g); gru(g); }
  }

  // ---- GraphNorm + readouts ----
  for (int g = 0; g < 3; ++g)
    k_graphnorm<<<BGR, 128, 0, stream>>>(xsP[g], gn_w, gn_b, gn_ms, xgP[g]);
  k_outs1<<<(BGR*512)/256, 256, 0, stream>>>(xgP[0], xgP[1], xgP[2], reac, prodS, outs1);
  for (int g = 0; g < 3; ++g){
    k_segmax <<<BGR, 128, 0, stream>>>(xgP[g], outs2, 384, g*128);
    k_segmean<<<BGR, 128, 0, stream>>>(xgP[g], outs3, 384, g*128);
    k_set2set<<<BGR, 128, 0, stream>>>(xgP[g], s2s_wih, s2s_whh, s2s_bih, s2s_bhh,
                                       outs4, 768, g*256);
  }

  // ---- head ----
  const float* oin[4] = {outs1, outs2, outs3, outs4};
  int oK[4] = {512, 384, 384, 768};
  for (int i = 0; i < 4; ++i){
    cvt(oin[i], o16, BGR*oK[i]);
    gemm(oK[i], o16, oK[i], blwp[i], bl_b[i], ycat + i*512, 2048, BGR, 512, 0);
    k_bnlin<<<(BGR*512)/256, 256, 0, stream>>>(ycat + i*512, 2048, bl_g[i], bl_be[i]);
  }
  cvt(ycat, o16, BGR*2048);
  gemm(2048, o16, 2048, ow1p, out_b1, y1, 512, BGR, 512, 0);
  k_prelu<<<(BGR*512)/256, 256, 0, stream>>>(y1, out_a1, BGR*512);
  cvt(y1, o16, BGR*512);
  gemm(512, o16, 512, ow2p, out_b2, y2, 512, BGR, 512, 0);
  k_prelu<<<(BGR*512)/256, 256, 0, stream>>>(y2, out_a2, BGR*512);
  k_final<<<BGR, 128, 0, stream>>>(y2, out_w3, out_b3, (float*)d_out);
}
